// MultiHeadAttention_11897059409944
// MI455X (gfx1250) — matrix operand order repaired and checked by reading, not compiled, not run on hardware
//
#include <hip/hip_runtime.h>
#include <hip/hip_bf16.h>

typedef __bf16 bf16_t;
typedef __attribute__((ext_vector_type(16))) __bf16 v16bf;
typedef __attribute__((ext_vector_type(8)))  __bf16 v8bf;
typedef __attribute__((ext_vector_type(8)))  float  v8f;
typedef __attribute__((ext_vector_type(4)))  float  v4f;

#define D_MODEL 1024
#define NHEADS  16
#define DK      64
#define SEQ     2048
#define BATCH   2

#define WMMA_BF16(a, b, c) \
  __builtin_amdgcn_wmma_f32_16x16x32_bf16(false, (a), false, (b), (short)0, (c), false, false)

// ---------- conversion helpers ----------
__device__ __forceinline__ v8bf cvt8v(v4f a, v4f b) {
  v8bf r;
  r[0] = (bf16_t)a[0]; r[1] = (bf16_t)a[1]; r[2] = (bf16_t)a[2]; r[3] = (bf16_t)a[3];
  r[4] = (bf16_t)b[0]; r[5] = (bf16_t)b[1]; r[6] = (bf16_t)b[2]; r[7] = (bf16_t)b[3];
  return r;
}
__device__ __forceinline__ v16bf cat16(v8bf lo, v8bf hi) {
  return __builtin_shufflevector(lo, hi, 0,1,2,3,4,5,6,7,8,9,10,11,12,13,14,15);
}
// A-frag (16x32): lane = row, elements 0..15 = K[kb..kb+15] (contiguous)
__device__ __forceinline__ v16bf load_a_bf16(const bf16_t* __restrict__ X, int ld, int m, int kb) {
  const bf16_t* p = X + (size_t)m * ld + kb;
  return cat16(*(const v8bf*)p, *(const v8bf*)(p + 8));
}
// B-frag (32x16): lane = column, elements 0..15 = K[kb..kb+15] (contiguous)
__device__ __forceinline__ v16bf load_b_bf16(const bf16_t* __restrict__ X, int ld, int n, int kb) {
  return *(const v16bf*)(X + (size_t)n * ld + kb);
}

// ---------- kernel 0: one-shot weight conversion f32 -> bf16 ----------
__global__ __launch_bounds__(256) void cvt_w_kernel(
    const float* __restrict__ Wq, const float* __restrict__ Wk,
    const float* __restrict__ Wv, const float* __restrict__ Wo,
    bf16_t* __restrict__ out)
{
  const int which = blockIdx.y;
  const float* W = (which == 0) ? Wq : (which == 1) ? Wk : (which == 2) ? Wv : Wo;
  const size_t i = ((size_t)blockIdx.x * 256 + threadIdx.x) * 8;   // 8 elems/thread
  v4f a = *(const v4f*)(W + i);
  v4f b = *(const v4f*)(W + i + 4);
  *(v8bf*)(out + (size_t)which * D_MODEL * D_MODEL + i) = cvt8v(a, b);
}

// ---------- prefetch helpers for the GEMM pipelines ----------
__device__ __forceinline__ void issue_proj_loads(
    const float* __restrict__ pa0, const float* __restrict__ pa1,
    const bf16_t* __restrict__ w0, const bf16_t* __restrict__ w1,
    const bf16_t* __restrict__ w2, const bf16_t* __restrict__ w3,
    int k0, v4f ra0[4], v4f ra1[4], v16bf rb[4])
{
  ra0[0] = *(const v4f*)(pa0 + k0);      ra0[1] = *(const v4f*)(pa0 + k0 + 4);
  ra0[2] = *(const v4f*)(pa0 + k0 + 8); ra0[3] = *(const v4f*)(pa0 + k0 + 12);
  ra1[0] = *(const v4f*)(pa1 + k0);      ra1[1] = *(const v4f*)(pa1 + k0 + 4);
  ra1[2] = *(const v4f*)(pa1 + k0 + 8); ra1[3] = *(const v4f*)(pa1 + k0 + 12);
  rb[0] = *(const v16bf*)(w0 + k0);
  rb[1] = *(const v16bf*)(w1 + k0);
  rb[2] = *(const v16bf*)(w2 + k0);
  rb[3] = *(const v16bf*)(w3 + k0);
}

__device__ __forceinline__ void issue_oproj_loads(
    const bf16_t* __restrict__ pa0, const bf16_t* __restrict__ pa1,
    const bf16_t* __restrict__ w0, const bf16_t* __restrict__ w1,
    const bf16_t* __restrict__ w2, const bf16_t* __restrict__ w3,
    int k0, v8bf ra[4], v16bf rb[4])
{
  ra[0] = *(const v8bf*)(pa0 + k0); ra[1] = *(const v8bf*)(pa0 + k0 + 8);
  ra[2] = *(const v8bf*)(pa1 + k0); ra[3] = *(const v8bf*)(pa1 + k0 + 8);
  rb[0] = *(const v16bf*)(w0 + k0);
  rb[1] = *(const v16bf*)(w1 + k0);
  rb[2] = *(const v16bf*)(w2 + k0);
  rb[3] = *(const v16bf*)(w3 + k0);
}

// ---------- kernel 1: fused QKV projection (32x64 tile per wave, pipelined) ----------
// Y[m,n] = sum_k X[m,k] * W[n,k] + b[n]; bf16 head-split (Q,K) / transposed (V)
__global__ __launch_bounds__(128) void proj_qkv_kernel(
    const float* __restrict__ q, const float* __restrict__ k, const float* __restrict__ v,
    const bf16_t* __restrict__ Wbf,
    const float* __restrict__ bq, const float* __restrict__ bk, const float* __restrict__ bv,
    bf16_t* __restrict__ Qh, bf16_t* __restrict__ Kh, bf16_t* __restrict__ Vt)
{
  const int wv   = threadIdx.x >> 5;
  const int lane = threadIdx.x & 31;
  const int half = lane >> 4;
  const int col  = lane & 15;
  const int m0 = (blockIdx.x * 4 + wv) * 32;        // 32 rows per wave
  const int n0 = blockIdx.y * 64;
  const int which = blockIdx.z;

  const float* X    = (which == 0) ? q  : (which == 1) ? k  : v;
  const float* bias = (which == 0) ? bq : (which == 1) ? bk : bv;
  const bf16_t* Wb  = Wbf + (size_t)which * D_MODEL * D_MODEL;

  const float*  pa0 = X  + (size_t)(m0 + col) * D_MODEL + 16 * half;
  const float*  pa1 = pa0 + (size_t)16 * D_MODEL;
  const bf16_t* wr0 = Wb + (size_t)(n0 +  0 + col) * D_MODEL + 16 * half;
  const bf16_t* wr1 = Wb + (size_t)(n0 + 16 + col) * D_MODEL + 16 * half;
  const bf16_t* wr2 = Wb + (size_t)(n0 + 32 + col) * D_MODEL + 16 * half;
  const bf16_t* wr3 = Wb + (size_t)(n0 + 48 + col) * D_MODEL + 16 * half;

  v4f ra0[4], ra1[4]; v16bf rb[4];
  issue_proj_loads(pa0, pa1, wr0, wr1, wr2, wr3, 0, ra0, ra1, rb);

  v8f acc[2][4] = {};
  #pragma unroll
  for (int k0 = 0; k0 < D_MODEL; k0 += 32) {
    v4f ca0[4], ca1[4]; v16bf cb[4];
    #pragma unroll
    for (int t = 0; t < 4; ++t) { ca0[t] = ra0[t]; ca1[t] = ra1[t]; cb[t] = rb[t]; }
    if (k0 + 32 < D_MODEL)
      issue_proj_loads(pa0, pa1, wr0, wr1, wr2, wr3, k0 + 32, ra0, ra1, rb);
    __builtin_amdgcn_sched_barrier(0);   // keep prefetch loads above the math
    v16bf a0 = cat16(cvt8v(ca0[0], ca0[1]), cvt8v(ca0[2], ca0[3]));
    v16bf a1 = cat16(cvt8v(ca1[0], ca1[1]), cvt8v(ca1[2], ca1[3]));
    #pragma unroll
    for (int j = 0; j < 4; ++j) {
      acc[0][j] = WMMA_BF16(a0, cb[j], acc[0][j]);
      acc[1][j] = WMMA_BF16(a1, cb[j], acc[1][j]);
    }
  }

  #pragma unroll
  for (int j = 0; j < 4; ++j) {
    const int n  = n0 + 16 * j + col;
    const float bb = bias[n];
    const int h = n >> 6, d = n & 63;
    #pragma unroll
    for (int i = 0; i < 2; ++i) {
      #pragma unroll
      for (int r = 0; r < 8; ++r) {
        const int srow = m0 + 16 * i + r + 8 * half;
        const int bidx = srow >> 11;
        const int sin  = srow & (SEQ - 1);
        const float val = acc[i][j][r] + bb;
        if (which == 2) {
          Vt[(((size_t)bidx * NHEADS + h) * DK + d) * SEQ + sin] = (bf16_t)val;
        } else {
          bf16_t* dst = (which == 0) ? Qh : Kh;
          dst[(((size_t)bidx * NHEADS + h) * SEQ + sin) * DK + d] = (bf16_t)val;
        }
      }
    }
  }
}

// ---------- kernel 2: flash attention (one wave per 16 query rows) ----------
__global__ __launch_bounds__(128) void attn_kernel(
    const bf16_t* __restrict__ Qh, const bf16_t* __restrict__ Kh,
    const bf16_t* __restrict__ Vt, const int* __restrict__ mask,
    bf16_t* __restrict__ Aout)
{
  __shared__ bf16_t plds[4][16 * 64];   // per-wave P staging (C-layout -> A-layout)
  const int wv   = threadIdx.x >> 5;
  const int lane = threadIdx.x & 31;
  const int half = lane >> 4;
  const int col  = lane & 15;
  const int bh = blockIdx.y;            // 0..31
  const int b = bh >> 4, h = bh & 15;
  const int m0 = (blockIdx.x * 4 + wv) * 16;

  const bf16_t* Qp = Qh + ((size_t)b * NHEADS + h) * SEQ * DK;
  const bf16_t* Kp = Kh + ((size_t)b * NHEADS + h) * SEQ * DK;
  const bf16_t* Vp = Vt + ((size_t)b * NHEADS + h) * DK * SEQ;
  const int*    mp = mask + (size_t)b * SEQ;

  const v16bf aq0 = load_a_bf16(Qp, DK, m0 + col, 0  + 16 * half);
  const v16bf aq1 = load_a_bf16(Qp, DK, m0 + col, 32 + 16 * half);

  v8f o[4] = {};
  float mrow[8], lrow[8];
  #pragma unroll
  for (int r = 0; r < 8; ++r) { mrow[r] = -3.0e38f; lrow[r] = 0.0f; }

  const float scale = 0.125f;                       // 1/sqrt(64)
  const float LOG2E = 1.4426950408889634f;
  bf16_t* myp = &plds[wv][0];

  for (int kv = 0; kv < SEQ; kv += 64) {
    // ---- batch-issue all K-fragments + mask for this 64-key chunk ----
    v16bf bk[4][2];
    int mk[4];
    #pragma unroll
    for (int j = 0; j < 4; ++j) {
      const int key = kv + 16 * j + col;
      bk[j][0] = load_b_bf16(Kp, DK, key, 0  + 16 * half);
      bk[j][1] = load_b_bf16(Kp, DK, key, 32 + 16 * half);
      mk[j] = mp[key];
    }
    __builtin_amdgcn_sched_barrier(0);
    // ---- QK^T: 8 WMMAs interleaved over 4 independent accumulators ----
    v8f s[4] = {};
    #pragma unroll
    for (int j = 0; j < 4; ++j) s[j] = WMMA_BF16(aq0, bk[j][0], s[j]);
    #pragma unroll
    for (int j = 0; j < 4; ++j) s[j] = WMMA_BF16(aq1, bk[j][1], s[j]);

    // ---- scale + mask + chunk row-max ----
    float p[4][8];
    float cmax[8];
    #pragma unroll
    for (int r = 0; r < 8; ++r) cmax[r] = -3.0e38f;
    #pragma unroll
    for (int j = 0; j < 4; ++j) {
      const bool dead = (mk[j] == 0);
      #pragma unroll
      for (int r = 0; r < 8; ++r) {
        float sv = dead ? -1.0e9f : s[j][r] * scale;
        p[j][r] = sv;
        cmax[r] = fmaxf(cmax[r], sv);
      }
    }
    #pragma unroll
    for (int r = 0; r < 8; ++r) {
      float vmax = cmax[r];
      vmax = fmaxf(vmax, __shfl_xor(vmax, 1));
      vmax = fmaxf(vmax, __shfl_xor(vmax, 2));
      vmax = fmaxf(vmax, __shfl_xor(vmax, 4));
      vmax = fmaxf(vmax, __shfl_xor(vmax, 8));
      cmax[r] = vmax;
    }

    // ---- online softmax update ----
    #pragma unroll
    for (int r = 0; r < 8; ++r) {
      const float mnew  = fmaxf(mrow[r], cmax[r]);
      const float alpha = exp2f((mrow[r] - mnew) * LOG2E);
      mrow[r] = mnew;
      float srow = 0.0f;
      #pragma unroll
      for (int j = 0; j < 4; ++j) {
        const float pv = exp2f((p[j][r] - mnew) * LOG2E);
        p[j][r] = pv;
        srow += pv;
      }
      srow += __shfl_xor(srow, 1);
      srow += __shfl_xor(srow, 2);
      srow += __shfl_xor(srow, 4);
      srow += __shfl_xor(srow, 8);
      lrow[r] = lrow[r] * alpha + srow;
      #pragma unroll
      for (int c = 0; c < 4; ++c) o[c][r] *= alpha;
    }

    // ---- issue V-fragment loads now: latency hides behind LDS round-trip ----
    v16bf bvv[4][2];
    #pragma unroll
    for (int c = 0; c < 4; ++c) {
      bvv[c][0] = load_b_bf16(Vp, SEQ, 16 * c + col, kv + 0  + 16 * half);
      bvv[c][1] = load_b_bf16(Vp, SEQ, 16 * c + col, kv + 32 + 16 * half);
    }
    __builtin_amdgcn_sched_barrier(0);

    // ---- P: C-layout regs -> LDS row-major -> A-layout (same-wave DS order) ----
    #pragma unroll
    for (int j = 0; j < 4; ++j)
      #pragma unroll
      for (int r = 0; r < 8; ++r)
        myp[(r + 8 * half) * 64 + 16 * j + col] = (bf16_t)p[j][r];

    const v16bf ap0 = load_a_bf16(myp, 64, col, 0  + 16 * half);
    const v16bf ap1 = load_a_bf16(myp, 64, col, 32 + 16 * half);

    // ---- O += P * V : 8 WMMAs interleaved over 4 accumulators ----
    #pragma unroll
    for (int c = 0; c < 4; ++c) o[c] = WMMA_BF16(ap0, bvv[c][0], o[c]);
    #pragma unroll
    for (int c = 0; c < 4; ++c) o[c] = WMMA_BF16(ap1, bvv[c][1], o[c]);
  }

  // normalize and write merged-head bf16 output [B,S,D]
  #pragma unroll
  for (int c = 0; c < 4; ++c) {
    #pragma unroll
    for (int r = 0; r < 8; ++r) {
      const int srow = m0 + r + 8 * half;
      const float val = o[c][r] / lrow[r];
      Aout[((size_t)b * SEQ + srow) * D_MODEL + h * DK + 16 * c + col] = (bf16_t)val;
    }
  }
}

// ---------- kernel 3: output projection (32x64 tile per wave, pipelined) ----------
__global__ __launch_bounds__(128) void out_proj_kernel(
    const bf16_t* __restrict__ Xa, const bf16_t* __restrict__ Wob,
    const float* __restrict__ bo, float* __restrict__ Y)
{
  const int wv   = threadIdx.x >> 5;
  const int lane = threadIdx.x & 31;
  const int half = lane >> 4;
  const int col  = lane & 15;
  const int m0 = (blockIdx.x * 4 + wv) * 32;
  const int n0 = blockIdx.y * 64;

  const bf16_t* pa0 = Xa  + (size_t)(m0 + col) * D_MODEL + 16 * half;
  const bf16_t* pa1 = pa0 + (size_t)16 * D_MODEL;
  const bf16_t* wr0 = Wob + (size_t)(n0 +  0 + col) * D_MODEL + 16 * half;
  const bf16_t* wr1 = Wob + (size_t)(n0 + 16 + col) * D_MODEL + 16 * half;
  const bf16_t* wr2 = Wob + (size_t)(n0 + 32 + col) * D_MODEL + 16 * half;
  const bf16_t* wr3 = Wob + (size_t)(n0 + 48 + col) * D_MODEL + 16 * half;

  v8bf ra[4]; v16bf rb[4];
  issue_oproj_loads(pa0, pa1, wr0, wr1, wr2, wr3, 0, ra, rb);

  v8f acc[2][4] = {};
  #pragma unroll
  for (int k0 = 0; k0 < D_MODEL; k0 += 32) {
    v8bf ca[4]; v16bf cb[4];
    #pragma unroll
    for (int t = 0; t < 4; ++t) { ca[t] = ra[t]; cb[t] = rb[t]; }
    if (k0 + 32 < D_MODEL)
      issue_oproj_loads(pa0, pa1, wr0, wr1, wr2, wr3, k0 + 32, ra, rb);
    __builtin_amdgcn_sched_barrier(0);
    v16bf a0 = cat16(ca[0], ca[1]);
    v16bf a1 = cat16(ca[2], ca[3]);
    #pragma unroll
    for (int j = 0; j < 4; ++j) {
      acc[0][j] = WMMA_BF16(a0, cb[j], acc[0][j]);
      acc[1][j] = WMMA_BF16(a1, cb[j], acc[1][j]);
    }
  }
  #pragma unroll
  for (int j = 0; j < 4; ++j) {
    const int n = n0 + 16 * j + col;
    const float bb = bo[n];
    #pragma unroll
    for (int i = 0; i < 2; ++i) {
      #pragma unroll
      for (int r = 0; r < 8; ++r) {
        const int srow = m0 + 16 * i + r + 8 * half;
        Y[(size_t)srow * D_MODEL + n] = acc[i][j][r] + bb;
      }
    }
  }
}

// ---------- host launch ----------
extern "C" void kernel_launch(void* const* d_in, const int* in_sizes, int n_in,
                              void* d_out, int out_size, void* d_ws, size_t ws_size,
                              hipStream_t stream) {
  const float* q    = (const float*)d_in[0];
  const float* k    = (const float*)d_in[1];
  const float* v    = (const float*)d_in[2];
  const int*   mask = (const int*)  d_in[3];
  const float* Wq   = (const float*)d_in[4];
  const float* bq   = (const float*)d_in[5];
  const float* Wk   = (const float*)d_in[6];
  const float* bk   = (const float*)d_in[7];
  const float* Wv   = (const float*)d_in[8];
  const float* bv   = (const float*)d_in[9];
  const float* Wo   = (const float*)d_in[10];
  const float* bo   = (const float*)d_in[11];

  const size_t HSZ = (size_t)BATCH * NHEADS * SEQ * DK;   // 4,194,304 elems
  const size_t WSZ = (size_t)D_MODEL * D_MODEL;           // 1,048,576 elems / matrix
  bf16_t* Qh   = (bf16_t*)d_ws;
  bf16_t* Kh   = Qh + HSZ;
  bf16_t* Vt   = Kh + HSZ;
  bf16_t* Aout = Vt + HSZ;
  bf16_t* Wbf  = Aout + HSZ;                              // 4 matrices, bf16

  // one-shot weight conversion
  cvt_w_kernel<<<dim3(512, 4), 256, 0, stream>>>(Wq, Wk, Wv, Wo, Wbf);

  // QKV projections: 4096/(32*4)=32 x 1024/64=16 x {Q,K,V}
  proj_qkv_kernel<<<dim3(32, 16, 3), 128, 0, stream>>>(
      q, k, v, Wbf, bq, bk, bv, Qh, Kh, Vt);

  // attention: 128 m-tiles / 4 waves = 32 blocks; 32 (b,h) pairs
  attn_kernel<<<dim3(32, 32), 128, 0, stream>>>(Qh, Kh, Vt, mask, Aout);

  // output projection -> f32 d_out
  out_proj_kernel<<<dim3(32, 16), 128, 0, stream>>>(Aout, Wbf + 3 * WSZ, bo, (float*)d_out);
}
